// MetalQuantizedLinear_2044404433312
// MI455X (gfx1250) — matrix operand order repaired and checked by reading, not compiled, not run on hardware
//
#include <hip/hip_runtime.h>

typedef __attribute__((ext_vector_type(16))) _Float16 v16h;
typedef __attribute__((ext_vector_type(8)))  _Float16 v8h;
typedef __attribute__((ext_vector_type(4)))  _Float16 v4h;
typedef __attribute__((ext_vector_type(2)))  _Float16 v2h;
typedef __attribute__((ext_vector_type(8)))  float    v8f;

#define BM 128
#define BN 128
#define BK 64
#define LDA (BK + 8)   // half elements; 144-byte row stride (16B aligned)
#define GQ 128         // quant group size along K

union AF { v16h v; v8h h[2]; };

// CDNA5 async copy: global -> LDS, 16 bytes per lane, tracked by ASYNCcnt.
// LDS aperture places the byte offset in addr[31:0], so truncating the flat
// address of a __shared__ element yields the LDS offset operand.
__device__ __forceinline__ void async_copy_b128(void* lds_dst, const void* gsrc) {
    const unsigned int       ldsoff = (unsigned int)(uintptr_t)lds_dst;
    const unsigned long long gaddr  = (unsigned long long)(uintptr_t)gsrc;
    asm volatile("global_load_async_to_lds_b128 %0, %1, off"
                 :: "v"(ldsoff), "v"(gaddr) : "memory");
}
__device__ __forceinline__ void wait_async_zero() {
    asm volatile("s_wait_asynccnt 0x0" ::: "memory");
}

// Dequantize one packed int32 (4 int8 along K) to 4 f16 values: (q - z) * s.
// Magic-number path: u = q+128 via XOR; 0x64 | u is f16(1024+u); 1024+u-1152 = q
// exactly in f16. Then one v_pk_fma_f16 per pair applies (q*s - z*s).
__device__ __forceinline__ v4h dequant4(int v, _Float16 s, _Float16 z) {
    union UH { unsigned int u; v2h h; };
    union V4 { v4h v; v2h h[2]; };

    const unsigned int ux = (unsigned int)v ^ 0x80808080u;
    UH lo, hi;
    lo.u = __builtin_amdgcn_perm(0x64646464u, ux, 0x04010400u); // f16: 1024+u0, 1024+u1
    hi.u = __builtin_amdgcn_perm(0x64646464u, ux, 0x04030402u); // f16: 1024+u2, 1024+u3

    const _Float16 k = (_Float16)1152;      // 1024 + 128 (exact in f16)
    const v2h k2   = { k, k };
    const v2h s2   = { s, s };
    const _Float16 nzs = (_Float16)0 - z * s;
    const v2h nzs2 = { nzs, nzs };

    const v2h qlo = lo.h - k2;              // exact integer in f16 (v_pk_add_f16)
    const v2h qhi = hi.h - k2;

    V4 r;
    r.h[0] = qlo * s2 + nzs2;               // v_pk_fma_f16
    r.h[1] = qhi * s2 + nzs2;
    return r.v;
}

__global__ __launch_bounds__(256, 1)
void qlinear_wmma_kernel(const _Float16* __restrict__ xg,
                         const int*      __restrict__ wp,
                         const _Float16* __restrict__ scalesp,
                         const _Float16* __restrict__ zerosp,
                         const float*    __restrict__ biasp,
                         float*          __restrict__ outp,
                         int M, int K, int N)
{
    __shared__ _Float16 sA [2][BM][LDA];   // x tile, [m][k]          (double buffered)
    __shared__ _Float16 sBt[2][BN][LDA];   // dequant W tile, [n][k]  (double buffered)

    const int tid  = threadIdx.x;
    const int lane = tid & 31;
    const int wave = tid >> 5;
    const int wm   = wave & 3;     // 4 waves along M: 4*32 = 128
    const int wn   = wave >> 2;    // 2 waves along N: 2*64 = 128
    const int lrow = lane & 15;
    const int lhi  = lane >> 4;

    const int m0 = blockIdx.y * BM;
    const int n0 = blockIdx.x * BN;

    v8f acc[2][4];
#pragma unroll
    for (int mi = 0; mi < 2; ++mi)
#pragma unroll
        for (int ni = 0; ni < 4; ++ni)
            acc[mi][ni] = 0.0f;

    // ---- prologue: tile 0 into buffer 0 ----
    {
#pragma unroll
        for (int i = 0; i < 4; ++i) {
            const int idx = tid + i * 256;
            const int row = idx >> 3;
            const int cg  = idx & 7;
            async_copy_b128(&sA[0][row][cg * 8],
                            xg + (size_t)(m0 + row) * K + cg * 8);
        }
#pragma unroll
        for (int i = 0; i < 2; ++i) {
            const int idx = tid + i * 256;
            const int r   = idx >> 5;
            const int c   = idx & 31;
            const int4 p  = *(const int4*)(wp + (size_t)r * N + n0 + c * 4);
            const v4h sv  = *(const v4h*)(scalesp + n0 + c * 4);
            const v4h zv  = *(const v4h*)(zerosp  + n0 + c * 4);
            const int comps[4] = { p.x, p.y, p.z, p.w };
#pragma unroll
            for (int j = 0; j < 4; ++j)
                *(v4h*)&sBt[0][c * 4 + j][r * 4] = dequant4(comps[j], sv[j], zv[j]);
        }
        wait_async_zero();
    }
    __syncthreads();

    const int nk = K / BK;
    for (int kt = 0; kt < nk; ++kt) {
        const int cur = kt & 1;
        const int nxt = cur ^ 1;
        const bool have_next = (kt + 1) < nk;

        // ---- next x tile: async DMA straight into LDS buffer `nxt`
        //      (copy engine runs while WMMAs execute on `cur`)            ----
        int4 wreg[2];
        v4h  sreg[2], zreg[2];
        if (have_next) {
            const int k1  = (kt + 1) * BK;
            const int kp1 = k1 >> 2;
            const int g1  = k1 / GQ;
#pragma unroll
            for (int i = 0; i < 4; ++i) {
                const int idx = tid + i * 256;
                const int row = idx >> 3;
                const int cg  = idx & 7;
                async_copy_b128(&sA[nxt][row][cg * 8],
                                xg + (size_t)(m0 + row) * K + k1 + cg * 8);
            }
            // next W tile: stage through VGPRs (dequant needs VALU anyway)
#pragma unroll
            for (int i = 0; i < 2; ++i) {
                const int idx = tid + i * 256;
                const int r   = idx >> 5;
                const int c   = idx & 31;
                wreg[i] = *(const int4*)(wp + (size_t)(kp1 + r) * N + n0 + c * 4);
                sreg[i] = *(const v4h*)(scalesp + (size_t)g1 * N + n0 + c * 4);
                zreg[i] = *(const v4h*)(zerosp  + (size_t)g1 * N + n0 + c * 4);
            }
            // warm L2 two tiles ahead (global_prefetch_b8)
            if (kt + 2 < nk) {
                const int k2 = (kt + 2) * BK;
                __builtin_prefetch(xg + (size_t)(m0 + (tid >> 1)) * K + k2 + (tid & 1) * 32, 0, 1);
                __builtin_prefetch(wp + (size_t)((k2 >> 2) + (tid >> 4)) * N + n0 + (tid & 15) * 8, 0, 1);
            }
        }

        // ---- compute on buffer `cur`: 2 k-steps x 8 WMMA ----
#pragma unroll
        for (int ks = 0; ks < 2; ++ks) {
            const int kk = ks * 32;

            v16h afrag[2];
#pragma unroll
            for (int mi = 0; mi < 2; ++mi) {
                const _Float16* ap = &sA[cur][wm * 32 + mi * 16 + lrow][kk + lhi * 8];
                AF ua;
                ua.h[0] = *(const v8h*)ap;          // K = kk + lhi*8 .. +7
                ua.h[1] = *(const v8h*)(ap + 16);   // K = kk+16 + lhi*8 .. +7
                afrag[mi] = ua.v;
            }

            v16h bfrag[4];
#pragma unroll
            for (int ni = 0; ni < 4; ++ni) {
                const _Float16* bp = &sBt[cur][wn * 64 + ni * 16 + lrow][kk + lhi * 8];
                AF ub;
                ub.h[0] = *(const v8h*)bp;
                ub.h[1] = *(const v8h*)(bp + 16);
                bfrag[ni] = ub.v;
            }

#pragma unroll
            for (int mi = 0; mi < 2; ++mi)
#pragma unroll
                for (int ni = 0; ni < 4; ++ni)
                    acc[mi][ni] = __builtin_amdgcn_wmma_f32_16x16x32_f16(
                        false, afrag[mi], false, bfrag[ni],
                        (short)0, acc[mi][ni], false, false);
        }

        // ---- commit next W tile to `nxt`; wait for async x copies ----
        if (have_next) {
#pragma unroll
            for (int i = 0; i < 2; ++i) {
                const int idx = tid + i * 256;
                const int r   = idx >> 5;
                const int c   = idx & 31;
                const int comps[4] = { wreg[i].x, wreg[i].y, wreg[i].z, wreg[i].w };
#pragma unroll
                for (int j = 0; j < 4; ++j)
                    *(v4h*)&sBt[nxt][c * 4 + j][r * 4] =
                        dequant4(comps[j], sreg[i][j], zreg[i][j]);
            }
            wait_async_zero();
        }
        __syncthreads();
    }

    // ---- epilogue: bias add + nontemporal f32 stores (streaming output) ----
#pragma unroll
    for (int mi = 0; mi < 2; ++mi) {
#pragma unroll
        for (int ni = 0; ni < 4; ++ni) {
            const int col = n0 + wn * 64 + ni * 16 + lrow;
            const float bsum = biasp[col];
#pragma unroll
            for (int r = 0; r < 8; ++r) {
                const int row = m0 + wm * 32 + mi * 16 + lhi * 8 + r;
                __builtin_nontemporal_store(acc[mi][ni][r] + bsum,
                                            outp + (size_t)row * N + col);
            }
        }
    }
}

extern "C" void kernel_launch(void* const* d_in, const int* in_sizes, int n_in,
                              void* d_out, int out_size, void* d_ws, size_t ws_size,
                              hipStream_t stream) {
    const _Float16* x      = (const _Float16*)d_in[0];
    const int*      wpk    = (const int*)d_in[1];
    const _Float16* scales = (const _Float16*)d_in[2];
    const _Float16* zeros  = (const _Float16*)d_in[3];
    const float*    bias   = (const float*)d_in[4];
    float*          out    = (float*)d_out;

    const int N = in_sizes[4];                       // 11008
    const int K = (in_sizes[2] / N) * GQ;            // 4096
    const int M = in_sizes[0] / K;                   // 4096

    dim3 grid(N / BN, M / BM);
    qlinear_wmma_kernel<<<grid, 256, 0, stream>>>(x, wpk, scales, zeros, bias, out,
                                                  M, K, N);
}
